// Seq2Seq_23553600651337
// MI455X (gfx1250) — matrix operand order repaired and checked by reading, not compiled, not run on hardware
//
#include <hip/hip_runtime.h>
#include <math.h>

// ---------------------------------------------------------------------------
// Types for WMMA (gfx1250, wave32): v_wmma_f32_16x16x32_f16
// ---------------------------------------------------------------------------
typedef _Float16 v16h __attribute__((ext_vector_type(16)));
typedef _Float16 v8h  __attribute__((ext_vector_type(8)));
typedef float    v8f  __attribute__((ext_vector_type(8)));

union F16x16 { v16h v; v8h h[2]; _Float16 e[16]; };

__device__ __forceinline__ v8f wmma16(v16h a, v16h b, v8f c) {
  return __builtin_amdgcn_wmma_f32_16x16x32_f16(
      /*neg_a=*/false, a, /*neg_b=*/false, b,
      /*c_mod=*/(short)0, c, /*reuse_a=*/false, /*reuse_b=*/false);
}

__device__ __forceinline__ v8f zero8() {
  v8f z;
#pragma unroll
  for (int e = 0; e < 8; ++e) z[e] = 0.f;
  return z;
}

// gfx1250 has a native v_tanh_f32 transcendental; use it when the builtin is
// exposed (1 TRANS op for tanh, 3 ops for sigmoid) instead of exp-based forms.
#if __has_builtin(__builtin_amdgcn_tanhf)
__device__ __forceinline__ float fast_tanh(float x) {
  return __builtin_amdgcn_tanhf(x);
}
__device__ __forceinline__ float fast_sigmoid(float x) {
  return __builtin_fmaf(0.5f, __builtin_amdgcn_tanhf(0.5f * x), 0.5f);
}
#else
__device__ __forceinline__ float fast_tanh(float x) {
  return 2.0f / (1.0f + __expf(-2.0f * x)) - 1.0f;  // saturates cleanly
}
__device__ __forceinline__ float fast_sigmoid(float x) {
  return 1.0f / (1.0f + __expf(-x));
}
#endif

// Load one B-fragment (32 lanes x 16 halves = one 32x16 f16 tile of W^T) from LDS.
__device__ __forceinline__ v16h load_bfrag(const _Float16* base, int frag, int lane) {
  const _Float16* p = base + frag * 512 + lane * 16;
  F16x16 r;
  r.h[0] = *(const v8h*)(p);
  r.h[1] = *(const v8h*)(p + 8);
  return r.v;
}

__device__ __forceinline__ void cp_lds_h(_Float16* dst, const _Float16* src,
                                         int nHalves, int tid) {
  const uint4* s = (const uint4*)src;
  uint4* d = (uint4*)dst;
  int n16 = nHalves >> 3;  // 8 halves per 16B
  for (int i = tid; i < n16; i += 256) d[i] = s[i];
}
__device__ __forceinline__ void cp_lds_f(float* dst, const float* src, int n, int tid) {
  for (int i = tid; i < n; i += 256) dst[i] = src[i];
}

// ---------------------------------------------------------------------------
// One LSTM cell step for a wave-owned 16-batch tile.
//   aX:  KCX input A-fragments (K = KCX*32, f16, zero-padded)
//   aH:  2 hidden A-fragments (K=64). Updated in place with new h.
//   wihF/whhF: B-fragments in LDS (NT=16 gate tiles). biasL: 256 f32 in LDS.
//   c[4]: cell state, C-layout (4 N-tiles of 16 hidden cols x 8 f32/lane).
//   stag: per-wave 16x64 f16 LDS staging for the C->A transpose of h.
// ---------------------------------------------------------------------------
template <int KCX>
__device__ __forceinline__ void lstm_step(const F16x16* aX, F16x16* aH,
                                          const _Float16* wihF,
                                          const _Float16* whhF,
                                          const float* biasL, v8f* c,
                                          _Float16* stag, int lane) {
  const int n = lane & 15;   // N column within tile / batch row on reload
  const int hi = lane >> 4;  // half-wave select
#pragma unroll
  for (int j = 0; j < 4; ++j) {  // 16-wide hidden slice
    v8f acc[4];
#pragma unroll
    for (int g = 0; g < 4; ++g) {  // i, f, g, o
      const int tile = j + 4 * g;
      v8f a = zero8();
#pragma unroll
      for (int kc = 0; kc < KCX; ++kc)
        a = wmma16(aX[kc].v, load_bfrag(wihF, tile * KCX + kc, lane), a);
#pragma unroll
      for (int kc = 0; kc < 2; ++kc)
        a = wmma16(aH[kc].v, load_bfrag(whhF, tile * 2 + kc, lane), a);
      const float bz = biasL[tile * 16 + n];
#pragma unroll
      for (int e = 0; e < 8; ++e) a[e] += bz;
      acc[g] = a;
    }
#pragma unroll
    for (int e = 0; e < 8; ++e) {
      const float iv = fast_sigmoid(acc[0][e]);
      const float fv = fast_sigmoid(acc[1][e]);
      const float gv = fast_tanh(acc[2][e]);
      const float ov = fast_sigmoid(acc[3][e]);
      const float cn = __builtin_fmaf(fv, c[j][e], iv * gv);
      c[j][e] = cn;
      const float hn = ov * fast_tanh(cn);
      // C-layout element (m = e + 8*hi, col = j*16+n) -> row-major staging
      stag[(e + 8 * hi) * 64 + j * 16 + n] = (_Float16)hn;
    }
  }
  // Intra-wave cross-lane LDS dependency: wait for stores before transpose read.
  asm volatile("s_wait_dscnt 0" ::: "memory");
#pragma unroll
  for (int kc = 0; kc < 2; ++kc) {
    const _Float16* p = stag + n * 64 + kc * 32 + hi * 8;  // A-frag layout
    aH[kc].h[0] = *(const v8h*)(p);
    aH[kc].h[1] = *(const v8h*)(p + 16);
  }
}

// ---------------------------------------------------------------------------
// Prep: pack f32 weight (NT*16 x Kreal) into wave32 WMMA B-fragments (f16).
// Fragment f = nt*KC + kc holds W^T tile: e<8 -> K=kc*32+8*(lane>>4)+e, e>=8 ->
// K=kc*32+16+8*(lane>>4)+(e-8), column n = lane&15; zero-padded past Kreal.
// ---------------------------------------------------------------------------
__global__ void pack_b_frags(const float* __restrict__ W, _Float16* __restrict__ dst,
                             int KC, int Kreal, int total) {
  int idx = blockIdx.x * blockDim.x + threadIdx.x;
  if (idx >= total) return;
  const int e = idx & 15;
  const int lane = (idx >> 4) & 31;
  const int fragid = idx >> 9;
  const int kc = fragid % KC;
  const int nt = fragid / KC;
  const int n = lane & 15;
  const int k = kc * 32 + ((e < 8) ? 0 : 16) + ((lane >> 4) ? 8 : 0) + (e & 7);
  float v = (k < Kreal) ? W[(nt * 16 + n) * Kreal + k] : 0.f;
  dst[idx] = (_Float16)v;
}

// ---------------------------------------------------------------------------
// Encoder: 2-layer LSTM over T=96. 8 waves/block, 16 batch rows/wave.
// Emits final h0, h1 as A-fragment blobs for the decoder.
// ---------------------------------------------------------------------------
__global__ void __launch_bounds__(256) enc_kernel(
    const float* __restrict__ X, const float* __restrict__ b0,
    const float* __restrict__ b1, const _Float16* __restrict__ wih0F,
    const _Float16* __restrict__ whh0F, const _Float16* __restrict__ wih1F,
    const _Float16* __restrict__ whh1F, _Float16* __restrict__ h0Out,
    _Float16* __restrict__ h1Out) {
  __shared__ __align__(16) _Float16 sWih0[16 * 512];
  __shared__ __align__(16) _Float16 sWhh0[32 * 512];
  __shared__ __align__(16) _Float16 sWih1[32 * 512];
  __shared__ __align__(16) _Float16 sWhh1[32 * 512];
  __shared__ float sB0[256];
  __shared__ float sB1[256];
  __shared__ __align__(16) _Float16 sStage[8][1024];

  const int tid = threadIdx.x;
  cp_lds_h(sWih0, wih0F, 16 * 512, tid);
  cp_lds_h(sWhh0, whh0F, 32 * 512, tid);
  cp_lds_h(sWih1, wih1F, 32 * 512, tid);
  cp_lds_h(sWhh1, whh1F, 32 * 512, tid);
  cp_lds_f(sB0, b0, 256, tid);
  cp_lds_f(sB1, b1, 256, tid);
  __syncthreads();

  const int wave = tid >> 5, lane = tid & 31;
  const int tileId = blockIdx.x * 8 + wave;
  const int n = lane & 15, hi = lane >> 4;
  _Float16* stag = sStage[wave];

  F16x16 aH0[2], aH1[2], aX[1];
#pragma unroll
  for (int kc = 0; kc < 2; ++kc)
#pragma unroll
    for (int e = 0; e < 16; ++e) { aH0[kc].e[e] = (_Float16)0; aH1[kc].e[e] = (_Float16)0; }
#pragma unroll
  for (int e = 0; e < 16; ++e) aX[0].e[e] = (_Float16)0;  // upper K half stays 0

  v8f c0[4], c1[4];
#pragma unroll
  for (int j = 0; j < 4; ++j) { c0[j] = zero8(); c1[j] = zero8(); }

  const float* xrow = X + ((size_t)(tileId * 16 + n) * 96) * 16 + hi * 8;

  // Double-buffered X: issue t+1's load before computing step t so the
  // global latency retires behind two full LSTM layer-steps.
  float4 nxa = *(const float4*)xrow;
  float4 nxb = *(const float4*)(xrow + 4);

  for (int t = 0; t < 96; ++t) {
    const float4 xa = nxa;
    const float4 xb = nxb;
    if (t < 95) {
      const float* xp = xrow + (t + 1) * 16;
      nxa = *(const float4*)xp;
      nxb = *(const float4*)(xp + 4);
    }
    aX[0].e[0] = (_Float16)xa.x; aX[0].e[1] = (_Float16)xa.y;
    aX[0].e[2] = (_Float16)xa.z; aX[0].e[3] = (_Float16)xa.w;
    aX[0].e[4] = (_Float16)xb.x; aX[0].e[5] = (_Float16)xb.y;
    aX[0].e[6] = (_Float16)xb.z; aX[0].e[7] = (_Float16)xb.w;

    lstm_step<1>(aX, aH0, sWih0, sWhh0, sB0, c0, stag, lane);   // layer 0
    lstm_step<2>(aH0, aH1, sWih1, sWhh1, sB1, c1, stag, lane);  // layer 1
  }

#pragma unroll
  for (int kc = 0; kc < 2; ++kc) {
    _Float16* p0 = h0Out + (size_t)tileId * 1024 + kc * 512 + lane * 16;
    *(v8h*)(p0) = aH0[kc].h[0];
    *(v8h*)(p0 + 8) = aH0[kc].h[1];
    _Float16* p1 = h1Out + (size_t)tileId * 1024 + kc * 512 + lane * 16;
    *(v8h*)(p1) = aH1[kc].h[0];
    *(v8h*)(p1 + 8) = aH1[kc].h[1];
  }
}

// ---------------------------------------------------------------------------
// Decoder: 24 autoregressive steps of 2 LSTM layers + MLP regressor head.
// ---------------------------------------------------------------------------
__global__ void __launch_bounds__(256) dec_kernel(
    const float* __restrict__ bias, const float* __restrict__ db0,
    const float* __restrict__ db1, const float* __restrict__ rb1,
    const float* __restrict__ rb2, const _Float16* __restrict__ wih0F,
    const _Float16* __restrict__ whh0F, const _Float16* __restrict__ wih1F,
    const _Float16* __restrict__ whh1F, const _Float16* __restrict__ w1F,
    const _Float16* __restrict__ w2F, const _Float16* __restrict__ h0In,
    const _Float16* __restrict__ h1In, float* __restrict__ out) {
  __shared__ __align__(16) _Float16 sWih0[32 * 512];
  __shared__ __align__(16) _Float16 sWhh0[32 * 512];
  __shared__ __align__(16) _Float16 sWih1[32 * 512];
  __shared__ __align__(16) _Float16 sWhh1[32 * 512];
  __shared__ __align__(16) _Float16 sW1[8 * 512];
  __shared__ __align__(16) _Float16 sW2[2 * 512];
  __shared__ float sDb0[256];
  __shared__ float sDb1[256];
  __shared__ float sRb1[64];
  __shared__ float sRb2[16];
  __shared__ __align__(16) _Float16 sStage[8][1024];

  const int tid = threadIdx.x;
  cp_lds_h(sWih0, wih0F, 32 * 512, tid);
  cp_lds_h(sWhh0, whh0F, 32 * 512, tid);
  cp_lds_h(sWih1, wih1F, 32 * 512, tid);
  cp_lds_h(sWhh1, whh1F, 32 * 512, tid);
  cp_lds_h(sW1, w1F, 8 * 512, tid);
  cp_lds_h(sW2, w2F, 2 * 512, tid);
  cp_lds_f(sDb0, db0, 256, tid);
  cp_lds_f(sDb1, db1, 256, tid);
  cp_lds_f(sRb1, rb1, 64, tid);
  cp_lds_f(sRb2, rb2, 16, tid);
  __syncthreads();

  const int wave = tid >> 5, lane = tid & 31;
  const int tileId = blockIdx.x * 8 + wave;
  const int n = lane & 15, hi = lane >> 4;
  const int mb = 8 * hi;
  _Float16* stag = sStage[wave];

  F16x16 aHt[2], aDh0[2], aDh1[2];
#pragma unroll
  for (int kc = 0; kc < 2; ++kc) {
    const _Float16* p0 = h0In + (size_t)tileId * 1024 + kc * 512 + lane * 16;
    aDh0[kc].h[0] = *(const v8h*)(p0);
    aDh0[kc].h[1] = *(const v8h*)(p0 + 8);
    const _Float16* p1 = h1In + (size_t)tileId * 1024 + kc * 512 + lane * 16;
    aHt[kc].h[0] = *(const v8h*)(p1);
    aHt[kc].h[1] = *(const v8h*)(p1 + 8);
    aDh1[kc] = aHt[kc];
  }
  v8f dc0[4], dc1[4];
#pragma unroll
  for (int j = 0; j < 4; ++j) { dc0[j] = zero8(); dc1[j] = zero8(); }

  float lb[8];
#pragma unroll
  for (int r = 0; r < 8; ++r)
    lb[r] = bias[(size_t)(tileId * 16 + mb + r) * 16 + n];
  const float b2n = sRb2[n];

  for (int s = 0; s < 24; ++s) {
    lstm_step<2>(aHt, aDh0, sWih0, sWhh0, sDb0, dc0, stag, lane);
    aHt[0] = aDh0[0]; aHt[1] = aDh0[1];
    lstm_step<2>(aHt, aDh1, sWih1, sWhh1, sDb1, dc1, stag, lane);
    aHt[0] = aDh1[0]; aHt[1] = aDh1[1];

    // Regressor: relu(h @ W1^T + b1) in C-layout, transpose, then @ W2^T.
#pragma unroll
    for (int j = 0; j < 4; ++j) {
      v8f r = zero8();
#pragma unroll
      for (int kc = 0; kc < 2; ++kc)
        r = wmma16(aHt[kc].v, load_bfrag(sW1, j * 2 + kc, lane), r);
      const float bb = sRb1[j * 16 + n];
#pragma unroll
      for (int e = 0; e < 8; ++e) {
        float v = r[e] + bb;
        v = v > 0.f ? v : 0.f;
        stag[(e + 8 * hi) * 64 + j * 16 + n] = (_Float16)v;
      }
    }
    asm volatile("s_wait_dscnt 0" ::: "memory");
    F16x16 aR[2];
#pragma unroll
    for (int kc = 0; kc < 2; ++kc) {
      const _Float16* p = stag + n * 64 + kc * 32 + hi * 8;
      aR[kc].h[0] = *(const v8h*)(p);
      aR[kc].h[1] = *(const v8h*)(p + 16);
    }
    v8f y = zero8();
#pragma unroll
    for (int kc = 0; kc < 2; ++kc)
      y = wmma16(aR[kc].v, load_bfrag(sW2, kc, lane), y);

    float* op = out + (size_t)(tileId * 16 + mb) * (24 * 16) + s * 16 + n;
#pragma unroll
    for (int r = 0; r < 8; ++r)
      op[(size_t)r * 24 * 16] = y[r] + b2n + lb[r];
  }
}

// ---------------------------------------------------------------------------
// Host: pack weights -> encoder -> decoder (all on `stream`).
// ---------------------------------------------------------------------------
extern "C" void kernel_launch(void* const* d_in, const int* in_sizes, int n_in,
                              void* d_out, int out_size, void* d_ws,
                              size_t ws_size, hipStream_t stream) {
  const float* X        = (const float*)d_in[0];
  const float* bias     = (const float*)d_in[1];
  // d_in[2] = X_mask (all ones, unused by the reference math)
  const float* enc_Wih0 = (const float*)d_in[3];
  const float* enc_Whh0 = (const float*)d_in[4];
  const float* enc_b0   = (const float*)d_in[5];
  const float* enc_Wih1 = (const float*)d_in[6];
  const float* enc_Whh1 = (const float*)d_in[7];
  const float* enc_b1   = (const float*)d_in[8];
  const float* dec_Wih0 = (const float*)d_in[9];
  const float* dec_Whh0 = (const float*)d_in[10];
  const float* dec_b0   = (const float*)d_in[11];
  const float* dec_Wih1 = (const float*)d_in[12];
  const float* dec_Whh1 = (const float*)d_in[13];
  const float* dec_b1   = (const float*)d_in[14];
  const float* reg_W1   = (const float*)d_in[15];
  const float* reg_b1   = (const float*)d_in[16];
  const float* reg_W2   = (const float*)d_in[17];
  const float* reg_b2   = (const float*)d_in[18];

  _Float16* ws = (_Float16*)d_ws;
  _Float16* encWih0F = ws + 0;                //  16 frags
  _Float16* encWhh0F = ws + 8192;             //  32 frags
  _Float16* encWih1F = ws + 24576;            //  32 frags
  _Float16* encWhh1F = ws + 40960;            //  32 frags
  _Float16* decWih0F = ws + 57344;
  _Float16* decWhh0F = ws + 73728;
  _Float16* decWih1F = ws + 90112;
  _Float16* decWhh1F = ws + 106496;
  _Float16* regW1F   = ws + 122880;           //   8 frags
  _Float16* regW2F   = ws + 126976;           //   2 frags
  _Float16* h0F      = ws + 128000;           // 1024 tiles x 1024 halves
  _Float16* h1F      = ws + 128000 + 1048576; // total ws need ~4.5 MB

  auto pack = [&](const float* W, _Float16* dst, int NT, int KC, int K) {
    const int total = NT * KC * 512;
    pack_b_frags<<<(total + 255) / 256, 256, 0, stream>>>(W, dst, KC, K, total);
  };
  pack(enc_Wih0, encWih0F, 16, 1, 16);
  pack(enc_Whh0, encWhh0F, 16, 2, 64);
  pack(enc_Wih1, encWih1F, 16, 2, 64);
  pack(enc_Whh1, encWhh1F, 16, 2, 64);
  pack(dec_Wih0, decWih0F, 16, 2, 64);
  pack(dec_Whh0, decWhh0F, 16, 2, 64);
  pack(dec_Wih1, decWih1F, 16, 2, 64);
  pack(dec_Whh1, decWhh1F, 16, 2, 64);
  pack(reg_W1, regW1F, 4, 2, 64);
  pack(reg_W2, regW2F, 1, 2, 64);

  enc_kernel<<<128, 256, 0, stream>>>(X, enc_b0, enc_b1, encWih0F, encWhh0F,
                                      encWih1F, encWhh1F, h0F, h1F);
  dec_kernel<<<128, 256, 0, stream>>>(bias, dec_b0, dec_b1, reg_b1, reg_b2,
                                      decWih0F, decWhh0F, decWih1F, decWhh1F,
                                      regW1F, regW2F, h0F, h1F, (float*)d_out);
}
